// WindowAttention_14697378086965
// MI455X (gfx1250) — matrix operand order repaired and checked by reading, not compiled, not run on hardware
//
#include <hip/hip_runtime.h>
#include <stdint.h>

// ---------------- problem constants ----------------
#define B_   4
#define T_   4096
#define C_   1024
#define H_   16
#define D_   64
#define W_   256
#define NW_  16
#define BT_  (B_ * T_)     // 16384
#define E_   (3 * C_)      // 3072

// ---------------- vector types ----------------
typedef __attribute__((ext_vector_type(16))) __bf16        v16bf;
typedef __attribute__((ext_vector_type(8)))  float         v8f;
typedef __attribute__((ext_vector_type(4)))  unsigned int  v4u;
typedef __attribute__((ext_vector_type(8)))  int           v8i;
typedef __attribute__((ext_vector_type(4)))  int           v4i;

union Frag { uint4 u[2]; v16bf v; };

__device__ __forceinline__ v8f wmma_bf16(const Frag& a, const Frag& b, v8f c) {
  // D = A(16x32 bf16) * B(32x16 bf16) + C(16x16 f32)
  return __builtin_amdgcn_wmma_f32_16x16x32_bf16(
      /*neg_a=*/false, a.v, /*neg_b=*/false, b.v,
      /*c_mod=*/(short)0, c, /*reuse_a=*/false, /*reuse_b=*/false);
}

__device__ __forceinline__ unsigned short f2bf(float x) {
  unsigned u = __builtin_bit_cast(unsigned, x);
  u += 0x7FFFu + ((u >> 16) & 1u);        // round-to-nearest-even
  return (unsigned short)(u >> 16);
}

// ---------------- fp32 -> bf16 pack (8 elems / thread) ----------------
__global__ __launch_bounds__(256) void cvt_f32_to_bf16(
    const float* __restrict__ src, unsigned short* __restrict__ dst, int n8) {
  int i = blockIdx.x * blockDim.x + threadIdx.x;
  if (i >= n8) return;
  const float4* s = reinterpret_cast<const float4*>(src) + (size_t)i * 2;
  float4 a = s[0], b = s[1];
  uint4 o;
  o.x = (unsigned)f2bf(a.x) | ((unsigned)f2bf(a.y) << 16);
  o.y = (unsigned)f2bf(a.z) | ((unsigned)f2bf(a.w) << 16);
  o.z = (unsigned)f2bf(b.x) | ((unsigned)f2bf(b.y) << 16);
  o.w = (unsigned)f2bf(b.z) | ((unsigned)f2bf(b.w) << 16);
  reinterpret_cast<uint4*>(dst)[i] = o;
}

// ---------------- QKV GEMM: [BT,C] x [E,C]^T, scatter to Q/K/Vt (bf16) ----
__global__ __launch_bounds__(256) void gemm_qkv(
    const unsigned short* __restrict__ A,   // xb  [BT_, C_] bf16 row-major
    const unsigned short* __restrict__ Wt,  // wab [E_,  C_] bf16 row-major
    unsigned short* __restrict__ qb,        // [B][nW][H][W][D]
    unsigned short* __restrict__ kb,        // [B][nW][H][W][D]
    unsigned short* __restrict__ vb) {      // [B][nW][H][D][W]  (transposed)
  const int lane = threadIdx.x & 31;
  const int wave = threadIdx.x >> 5;
  const int wm = wave >> 1, wn = wave & 1;
  const int lr = lane & 15, lh = lane >> 4;
  const int koff = lh * 8;

  const int m0 = blockIdx.x * 256 + wm * 64;
  const int n0 = blockIdx.y * 128 + wn * 64;

  const v8f vzero = {0.f,0.f,0.f,0.f,0.f,0.f,0.f,0.f};
  v8f acc[4][4];
  #pragma unroll
  for (int i = 0; i < 4; ++i)
    #pragma unroll
    for (int j = 0; j < 4; ++j) acc[i][j] = vzero;

  for (int k0 = 0; k0 < C_; k0 += 32) {
    Frag af[4], bf[4];
    #pragma unroll
    for (int i = 0; i < 4; ++i) {           // A frag: row=lr, K = koff..+7, 16+koff..+7
      const unsigned short* ap = A + (size_t)(m0 + i * 16 + lr) * C_ + k0 + koff;
      af[i].u[0] = *reinterpret_cast<const uint4*>(ap);
      af[i].u[1] = *reinterpret_cast<const uint4*>(ap + 16);
    }
    #pragma unroll
    for (int j = 0; j < 4; ++j) {           // B frag: col=lr, K = koff..+7, 16+koff..+7
      const unsigned short* bp = Wt + (size_t)(n0 + j * 16 + lr) * C_ + k0 + koff;
      bf[j].u[0] = *reinterpret_cast<const uint4*>(bp);
      bf[j].u[1] = *reinterpret_cast<const uint4*>(bp + 16);
    }
    #pragma unroll
    for (int i = 0; i < 4; ++i)
      #pragma unroll
      for (int j = 0; j < 4; ++j)
        acc[i][j] = wmma_bf16(af[i], bf[j], acc[i][j]);
  }

  // epilogue: e = three*1024 + h*64 + d ; token m -> (b, nw, i)
  #pragma unroll
  for (int j = 0; j < 4; ++j) {
    const int n = n0 + j * 16 + lr;
    const int three = n >> 10;
    const int h = (n & 1023) >> 6;
    const int d = n & 63;
    #pragma unroll
    for (int i = 0; i < 4; ++i) {
      #pragma unroll
      for (int r = 0; r < 8; ++r) {
        const int m = m0 + i * 16 + lh * 8 + r;
        const int b  = m >> 12;
        const int t  = m & 4095;
        const int nw = t >> 8;
        const int ii = t & 255;
        const size_t head = (size_t)(((b * NW_ + nw) * H_ + h)) * (W_ * D_);
        const unsigned short val = f2bf(acc[i][j][r]);
        if (three == 0)      qb[head + (size_t)ii * D_ + d] = val;
        else if (three == 1) kb[head + (size_t)ii * D_ + d] = val;
        else                 vb[head + (size_t)d * W_ + ii] = val;
      }
    }
  }
}

// ---------------- TDM: 1-D bf16 tile, global -> LDS ----------------
__device__ __forceinline__ void tdm_load_1d(unsigned lds_off, const void* gptr,
                                            unsigned nelem) {
  const unsigned long long ga = (unsigned long long)(uintptr_t)gptr;
  v4u g0;
  g0.x = 1u;                                         // count=1, user mode
  g0.y = lds_off;                                    // LDS byte address
  g0.z = (unsigned)(ga & 0xffffffffu);               // global_addr[31:0]
  g0.w = (unsigned)((ga >> 32) & 0x1ffffffu) | (2u << 30);  // addr[56:32] | type=2
  v8i g1;
  g1[0] = (int)(1u << 16);                           // data_size = 2 bytes
  g1[1] = (int)(nelem << 16);                        // tensor_dim0[15:0] -> bits[63:48]
  g1[2] = (int)(((nelem >> 16) & 0xffffu) | (1u << 16)); // dim0 hi | tensor_dim1=1 lo
  g1[3] = (int)(nelem << 16);                        // tile_dim0 -> bits[127:112]
  g1[4] = 1;                                         // tile_dim1 = 1
  g1[5] = (int)nelem;                                // tensor_dim0_stride lo32
  g1[6] = 0;
  g1[7] = 0;
  const v4i z4 = {0, 0, 0, 0};
  const v8i z8 = {0, 0, 0, 0, 0, 0, 0, 0};
  __builtin_amdgcn_tensor_load_to_lds(g0, g1, z4, z4, z8, 0);
}

// ---------------- flash window attention, one (b,window,head) / block ------
__global__ __launch_bounds__(512) void attn_win(
    const unsigned short* __restrict__ qg,   // [B][nW][H][W][D]
    const unsigned short* __restrict__ kg,   // [B][nW][H][W][D]
    const unsigned short* __restrict__ vg,   // [B][nW][H][D][W]
    unsigned short* __restrict__ yg) {       // [BT_, C_] bf16
  __shared__ __align__(16) unsigned short sK[W_ * D_];   // 32 KB: K rows [j][d]
  __shared__ __align__(16) unsigned short sV[D_ * W_];   // 32 KB: V^T  [d][j]
  __shared__ __align__(16) unsigned short sP[16 * 16 * 32]; // 16 waves x 1KB stage

  const unsigned flat = blockIdx.x;          // ((b*nW + nw)*H + h)
  const unsigned h  = flat & 15;
  const unsigned nw = (flat >> 4) & 15;
  const unsigned b  = flat >> 8;
  const unsigned short* Qh = qg + (size_t)flat * (W_ * D_);
  const unsigned short* Kh = kg + (size_t)flat * (W_ * D_);
  const unsigned short* Vh = vg + (size_t)flat * (W_ * D_);

  const int tid  = threadIdx.x;
  const int wave = tid >> 5;
  const int lane = tid & 31;
  const int lr = lane & 15, lh = lane >> 4;
  const int koff = lh * 8;

  if (tid < 32) {   // wave 0 issues both DMAs; TDM ignores EXEC
    tdm_load_1d((unsigned)(uintptr_t)sK, Kh, W_ * D_);
    tdm_load_1d((unsigned)(uintptr_t)sV, Vh, W_ * D_);
    __builtin_amdgcn_s_wait_tensorcnt(0);
  }
  __syncthreads();

  // Q fragments for this wave's 16 rows (kept in registers across j loop)
  const int ibase = wave * 16;
  Frag qf[2];
  {
    const unsigned short* qrow = Qh + (size_t)(ibase + lr) * D_;
    #pragma unroll
    for (int t = 0; t < 2; ++t) {
      qf[t].u[0] = *reinterpret_cast<const uint4*>(qrow + t * 32 + koff);
      qf[t].u[1] = *reinterpret_cast<const uint4*>(qrow + t * 32 + 16 + koff);
    }
  }

  const v8f vzero = {0.f,0.f,0.f,0.f,0.f,0.f,0.f,0.f};
  v8f yacc[4];
  #pragma unroll
  for (int dt = 0; dt < 4; ++dt) yacc[dt] = vzero;
  float mrun[8], lrun[8];
  #pragma unroll
  for (int r = 0; r < 8; ++r) { mrun[r] = -3.0e38f; lrun[r] = 0.f; }

  const float scale = 0.125f;  // 1/sqrt(64)
  unsigned short* myP = sP + wave * 512;

  for (int jb = 0; jb < W_; jb += 32) {
    // ---- S = Q K^T for 16 x 32 block (two 16x16 tiles) ----
    v8f s0 = vzero, s1 = vzero;
    #pragma unroll
    for (int t = 0; t < 2; ++t) {
      Frag kf;
      const unsigned short* kp = sK + (size_t)(jb + lr) * D_ + t * 32 + koff;
      kf.u[0] = *reinterpret_cast<const uint4*>(kp);
      kf.u[1] = *reinterpret_cast<const uint4*>(kp + 16);
      s0 = wmma_bf16(qf[t], kf, s0);
      kp = sK + (size_t)(jb + 16 + lr) * D_ + t * 32 + koff;
      kf.u[0] = *reinterpret_cast<const uint4*>(kp);
      kf.u[1] = *reinterpret_cast<const uint4*>(kp + 16);
      s1 = wmma_bf16(qf[t], kf, s1);
    }

    // ---- online softmax over the 32 new columns ----
    float corr[8];
    #pragma unroll
    for (int r = 0; r < 8; ++r) {
      float a = s0[r] * scale, c = s1[r] * scale;
      float mx = fmaxf(a, c);
      #pragma unroll
      for (int off = 1; off < 16; off <<= 1) mx = fmaxf(mx, __shfl_xor(mx, off, 16));
      const float mnew = fmaxf(mrun[r], mx);
      corr[r] = __expf(mrun[r] - mnew);
      const float p0 = __expf(a - mnew);
      const float p1 = __expf(c - mnew);
      float sm = p0 + p1;
      #pragma unroll
      for (int off = 1; off < 16; off <<= 1) sm += __shfl_xor(sm, off, 16);
      lrun[r] = lrun[r] * corr[r] + sm;
      mrun[r] = mnew;
      // stage P (bf16) row-major 16x32 for in-wave transpose
      myP[(size_t)(lh * 8 + r) * 32 + lr]      = f2bf(p0);
      myP[(size_t)(lh * 8 + r) * 32 + 16 + lr] = f2bf(p1);
    }
    #pragma unroll
    for (int dt = 0; dt < 4; ++dt)
      #pragma unroll
      for (int r = 0; r < 8; ++r) yacc[dt][r] *= corr[r];

    asm volatile("s_wait_dscnt 0" ::: "memory");

    // ---- reload P in A-fragment layout (row=lr, K=j local) ----
    Frag pf;
    const unsigned short* pp = myP + (size_t)lr * 32 + koff;
    pf.u[0] = *reinterpret_cast<const uint4*>(pp);
    pf.u[1] = *reinterpret_cast<const uint4*>(pp + 16);

    // ---- Y += P * V : 4 d-tiles, B frags from transposed V in LDS ----
    #pragma unroll
    for (int dt = 0; dt < 4; ++dt) {
      Frag vf;
      const unsigned short* vp = sV + (size_t)(dt * 16 + lr) * W_ + jb + koff;
      vf.u[0] = *reinterpret_cast<const uint4*>(vp);
      vf.u[1] = *reinterpret_cast<const uint4*>(vp + 16);
      yacc[dt] = wmma_bf16(pf, vf, yacc[dt]);
    }
  }

  // ---- normalize and store Y as bf16 into [BT_, C_] ----
  #pragma unroll
  for (int r = 0; r < 8; ++r) {
    const float rinv = 1.0f / lrun[r];
    const int ii = ibase + lh * 8 + r;                 // row in window
    const size_t gt = (size_t)b * T_ + (size_t)nw * W_ + ii;
    #pragma unroll
    for (int dt = 0; dt < 4; ++dt) {
      const int cc = h * D_ + dt * 16 + lr;
      yg[gt * C_ + cc] = f2bf(yacc[dt][r] * rinv);
    }
  }
}

// ---------------- Proj GEMM: [BT,C] x [C,C]^T -> f32 out ----------------
__global__ __launch_bounds__(256) void gemm_proj(
    const unsigned short* __restrict__ A,   // yb  [BT_, C_] bf16
    const unsigned short* __restrict__ Wt,  // wpb [C_,  C_] bf16
    float* __restrict__ out) {              // [BT_, C_] f32
  const int lane = threadIdx.x & 31;
  const int wave = threadIdx.x >> 5;
  const int wm = wave >> 1, wn = wave & 1;
  const int lr = lane & 15, lh = lane >> 4;
  const int koff = lh * 8;

  const int m0 = blockIdx.x * 256 + wm * 64;
  const int n0 = blockIdx.y * 128 + wn * 64;

  const v8f vzero = {0.f,0.f,0.f,0.f,0.f,0.f,0.f,0.f};
  v8f acc[4][4];
  #pragma unroll
  for (int i = 0; i < 4; ++i)
    #pragma unroll
    for (int j = 0; j < 4; ++j) acc[i][j] = vzero;

  for (int k0 = 0; k0 < C_; k0 += 32) {
    Frag af[4], bf[4];
    #pragma unroll
    for (int i = 0; i < 4; ++i) {
      const unsigned short* ap = A + (size_t)(m0 + i * 16 + lr) * C_ + k0 + koff;
      af[i].u[0] = *reinterpret_cast<const uint4*>(ap);
      af[i].u[1] = *reinterpret_cast<const uint4*>(ap + 16);
    }
    #pragma unroll
    for (int j = 0; j < 4; ++j) {
      const unsigned short* bp = Wt + (size_t)(n0 + j * 16 + lr) * C_ + k0 + koff;
      bf[j].u[0] = *reinterpret_cast<const uint4*>(bp);
      bf[j].u[1] = *reinterpret_cast<const uint4*>(bp + 16);
    }
    #pragma unroll
    for (int i = 0; i < 4; ++i)
      #pragma unroll
      for (int j = 0; j < 4; ++j)
        acc[i][j] = wmma_bf16(af[i], bf[j], acc[i][j]);
  }

  #pragma unroll
  for (int i = 0; i < 4; ++i)
    #pragma unroll
    for (int j = 0; j < 4; ++j) {
      const int n = n0 + j * 16 + lr;
      #pragma unroll
      for (int r = 0; r < 8; ++r) {
        const int m = m0 + i * 16 + lh * 8 + r;
        out[(size_t)m * C_ + n] = acc[i][j][r];
      }
    }
}

// ---------------- host-side launcher ----------------
extern "C" void kernel_launch(void* const* d_in, const int* in_sizes, int n_in,
                              void* d_out, int out_size, void* d_ws, size_t ws_size,
                              hipStream_t stream) {
  const float* x      = (const float*)d_in[0];   // [B,T,C]
  const float* w_attn = (const float*)d_in[1];   // [3C, C]
  const float* w_proj = (const float*)d_in[2];   // [C, C]
  float* out = (float*)d_out;

  char* ws = (char*)d_ws;
  const size_t XB  = 0;                               // 32 MiB
  const size_t WAB = XB  + (size_t)BT_ * C_ * 2;      //  6 MiB
  const size_t WPB = WAB + (size_t)E_  * C_ * 2;      //  2 MiB
  const size_t QB  = WPB + (size_t)C_  * C_ * 2;      // 32 MiB
  const size_t KB  = QB  + (size_t)BT_ * C_ * 2;      // 32 MiB
  const size_t VB  = KB  + (size_t)BT_ * C_ * 2;      // 32 MiB
  const size_t YB  = VB  + (size_t)BT_ * C_ * 2;      // 32 MiB
  unsigned short* xb  = (unsigned short*)(ws + XB);
  unsigned short* wab = (unsigned short*)(ws + WAB);
  unsigned short* wpb = (unsigned short*)(ws + WPB);
  unsigned short* qb  = (unsigned short*)(ws + QB);
  unsigned short* kb  = (unsigned short*)(ws + KB);
  unsigned short* vb  = (unsigned short*)(ws + VB);
  unsigned short* yb  = (unsigned short*)(ws + YB);

  // 1) demote fp32 -> bf16
  {
    int n8 = (BT_ * C_) / 8;           // 2,097,152
    cvt_f32_to_bf16<<<(n8 + 255) / 256, 256, 0, stream>>>(x, xb, n8);
    n8 = (E_ * C_) / 8;                //   393,216
    cvt_f32_to_bf16<<<(n8 + 255) / 256, 256, 0, stream>>>(w_attn, wab, n8);
    n8 = (C_ * C_) / 8;                //   131,072
    cvt_f32_to_bf16<<<(n8 + 255) / 256, 256, 0, stream>>>(w_proj, wpb, n8);
  }

  // 2) QKV projection + scatter to attention layouts
  gemm_qkv<<<dim3(BT_ / 256, E_ / 128), 256, 0, stream>>>(xb, wab, qb, kb, vb);

  // 3) windowed flash attention (one block per (b, window, head))
  attn_win<<<B_ * NW_ * H_, 512, 0, stream>>>(qb, kb, vb, yb);

  // 4) output projection (f32 result)
  gemm_proj<<<dim3(BT_ / 256, C_ / 128), 256, 0, stream>>>(yb, wpb, out);
}
